// EbdGNN_90752658964688
// MI455X (gfx1250) — matrix operand order repaired and checked by reading, not compiled, not run on hardware
//
// EbdGNN forward for MI455X (gfx1250): fused-weight bf16 WMMA GEMMs + CSR-gather
// aggregation (no feature atomics).
//
// Roofline: edge gather traffic (~1.2 GB, L2-resident) dominates; the naive
// scatter formulation needs ~307M fp32 L2 atomics (atomic-unit bound, >>100us),
// so we build a CSR (3.2M int atomics + one LDS prefix scan) and aggregate by
// gather: one wave per node, sum-in-registers, single store. GEMMs (21 GFLOP)
// run on v_wmma_f32_16x16x32_bf16 with h@Wg1 algebraically folded into a single
// K=640 input GEMM, so x1 (205 MB) is read exactly once via LDS-staged bf16
// conversion. BN+ReLU fuses into the layer-1 gather; bg2-init fuses into the
// layer-2 gather.

#include <hip/hip_runtime.h>
#include <hip/hip_bf16.h>
#include <stdint.h>
#include <stddef.h>

typedef __attribute__((ext_vector_type(16))) __bf16 v16bf;
typedef __attribute__((ext_vector_type(8)))  __bf16 v8bf;
typedef __attribute__((ext_vector_type(8)))  float  v8f;

#define NN    100000
#define EE    1600000
#define F1D   512
#define F3D   128
#define HD    128
#define CD    64
#define KTOT  640            // F1D + F3D (fused K)
#define FWC   0.8f
#define SWC   0.2f
#define BNEPS 1e-5f
#define LDA   40             // padded LDS row stride (bf16 elems) for A tiles

// Build a 16x32 bf16 A-fragment for one lane (contiguous K order, matching B):
// lanes 0-15 hold K=0..15, lanes 16-31 hold K=16..31;
// caller passes p already offset by (lane>=16 ? 16 : 0) elements.
__device__ __forceinline__ v16bf load_a_frag(const __bf16* p) {
  union { v16bf v; v8bf h[2]; } u;
  u.h[0] = *reinterpret_cast<const v8bf*>(p);        // K  +0..7
  u.h[1] = *reinterpret_cast<const v8bf*>(p + 8);    // K  +8..15
  return u.v;
}

// ---------------------------------------------------------------- prep
// BtZ[n][k] = bf16( scale(k) * (Wcat @ Wg1)[k][n] ), k in [0,640): fused weight,
// transposed so B-fragments are contiguous 32B loads. bz[n] = (fw*b1+sw*b2)@Wg1.
// Bt2[c][k] = bf16(Wg2[k][c]).
__global__ __launch_bounds__(256) void prep_kernel(
    const float* __restrict__ W1, const float* __restrict__ b1,
    const float* __restrict__ W2, const float* __restrict__ b2,
    const float* __restrict__ Wg1, const float* __restrict__ Wg2,
    __bf16* __restrict__ BtZ, __bf16* __restrict__ Bt2, float* __restrict__ bz) {
  const int i = blockIdx.x * 256 + threadIdx.x;
  if (i < KTOT * HD) {
    const int n = i / KTOT;
    const int k = i % KTOT;
    float s;
    const float* wrow;
    if (k < F1D) { s = FWC; wrow = W1 + (size_t)k * HD; }
    else         { s = SWC; wrow = W2 + (size_t)(k - F1D) * HD; }
    float acc = 0.f;
    for (int j = 0; j < HD; ++j) acc += wrow[j] * Wg1[j * HD + n];
    BtZ[(size_t)n * KTOT + k] = (__bf16)(s * acc);
  } else if (i < KTOT * HD + HD) {
    const int n = i - KTOT * HD;
    float acc = 0.f;
    for (int j = 0; j < HD; ++j) acc += (FWC * b1[j] + SWC * b2[j]) * Wg1[j * HD + n];
    bz[n] = acc;
  } else if (i < KTOT * HD + HD + CD * HD) {
    const int i3 = i - (KTOT * HD + HD);
    const int c = i3 / HD;
    const int k = i3 % HD;
    Bt2[(size_t)c * HD + k] = (__bf16)Wg2[(size_t)k * CD + c];
  }
}

// ---------------------------------------------------------------- CSR build
__global__ __launch_bounds__(256) void zero_deg_kernel(int* __restrict__ deg) {
  const int i = blockIdx.x * 256 + threadIdx.x;
  if (i < NN) deg[i] = 0;
}

__global__ __launch_bounds__(256) void deg_kernel(
    const int* __restrict__ ei, int* __restrict__ deg) {
  const int e = blockIdx.x * 256 + threadIdx.x;
  if (e < EE) atomicAdd(&deg[ei[EE + e]], 1);
}

// Single-workgroup exclusive prefix scan over deg[NN] -> rowStart / cursor.
__global__ __launch_bounds__(1024) void scan_kernel(
    const int* __restrict__ deg, int* __restrict__ rowStart, int* __restrict__ cursor) {
  __shared__ int buf[1024];
  __shared__ int carryS;
  const int tid = threadIdx.x;
  int carry = 0;
  for (int base = 0; base < NN; base += 1024) {
    const int idx = base + tid;
    const int v = (idx < NN) ? deg[idx] : 0;
    buf[tid] = v;
    __syncthreads();
#pragma unroll
    for (int off = 1; off < 1024; off <<= 1) {
      const int t = (tid >= off) ? buf[tid - off] : 0;
      __syncthreads();
      buf[tid] += t;
      __syncthreads();
    }
    const int incl = buf[tid];
    if (idx < NN) {
      const int excl = carry + incl - v;
      rowStart[idx] = excl;
      cursor[idx]   = excl;
    }
    if (tid == 1023) carryS = carry + buf[1023];
    __syncthreads();
    carry = carryS;
    __syncthreads();
  }
  if (tid == 0) rowStart[NN] = carry;   // == EE
}

// csrSrc[slot] = src node id, bucketed by dst via cursor bump.
__global__ __launch_bounds__(256) void csr_fill_kernel(
    const int* __restrict__ ei, int* __restrict__ cursor, int* __restrict__ csrSrc) {
  const int e = blockIdx.x * 256 + threadIdx.x;
  if (e >= EE) return;
  const int d = ei[EE + e];
  const int slot = atomicAdd(&cursor[d], 1);
  csrSrc[slot] = ei[e];
}

// ---------------------------------------------------------------- GEMM1
// z[N,128] = [x1 | x2] @ BtZ^T + bz  via bf16 WMMA, fp32 accumulate.
// Block: 256 thr = 8 waves as 2(row)x4(col); tile = 64 rows x 128 cols; K=640.
__global__ __launch_bounds__(256) void gemm1_kernel(
    const float* __restrict__ x1, const float* __restrict__ x2,
    const __bf16* __restrict__ BtZ, const float* __restrict__ bz,
    float* __restrict__ z) {
  __shared__ __attribute__((aligned(32))) __bf16 As[64 * LDA];

  const int tid  = threadIdx.x;
  const int lane = tid & 31;
  const int w    = tid >> 5;
  const int wy   = w & 1;        // 0..1 -> 32-row group
  const int wx   = w >> 1;       // 0..3 -> 32-col group
  const int half = lane >> 4;    // 0/1
  const int l16  = lane & 15;
  const int mBlock = blockIdx.x * 64;

  v8f acc[2][2];
#pragma unroll
  for (int mi = 0; mi < 2; ++mi)
#pragma unroll
    for (int ni = 0; ni < 2; ++ni)
      acc[mi][ni] = (v8f){0.f, 0.f, 0.f, 0.f, 0.f, 0.f, 0.f, 0.f};

  // staging role: each thread converts 8 fp32 -> bf16 per K-step
  const int srow = tid >> 2;          // 0..63
  const int skl  = (tid & 3) * 8;     // 0,8,16,24
  int gRow = mBlock + srow;
  if (gRow >= NN) gRow = NN - 1;      // clamp (dup read; stores are guarded)
  const float* rowp1 = x1 + (size_t)gRow * F1D;
  const float* rowp3 = x2 + (size_t)gRow * F3D;

  for (int kk = 0; kk < KTOT; kk += 32) {
    const float* src = (kk < F1D) ? (rowp1 + kk + skl) : (rowp3 + (kk - F1D) + skl);
    const float4 f0 = *(const float4*)(src);
    const float4 f1 = *(const float4*)(src + 4);
    if (kk + 32 < KTOT) __builtin_prefetch(src + 32, 0, 0);   // global_prefetch_b8
    __bf16* d = &As[srow * LDA + skl];
    d[0] = (__bf16)f0.x; d[1] = (__bf16)f0.y; d[2] = (__bf16)f0.z; d[3] = (__bf16)f0.w;
    d[4] = (__bf16)f1.x; d[5] = (__bf16)f1.y; d[6] = (__bf16)f1.z; d[7] = (__bf16)f1.w;
    __syncthreads();

    v16bf a[2], b[2];
#pragma unroll
    for (int mi = 0; mi < 2; ++mi) {
      const __bf16* p = &As[(wy * 32 + mi * 16 + l16) * LDA + half * 16];
      a[mi] = load_a_frag(p);
    }
#pragma unroll
    for (int ni = 0; ni < 2; ++ni) {
      // B 32x16: lanes 0-15 hold K 0..15, lanes 16-31 hold K 16..31 (contig 32B)
      const __bf16* q = BtZ + (size_t)(wx * 32 + ni * 16 + l16) * KTOT + kk + half * 16;
      b[ni] = *reinterpret_cast<const v16bf*>(q);
    }
#pragma unroll
    for (int mi = 0; mi < 2; ++mi)
#pragma unroll
      for (int ni = 0; ni < 2; ++ni)
        acc[mi][ni] = __builtin_amdgcn_wmma_f32_16x16x32_bf16(
            false, a[mi], false, b[ni], (short)0, acc[mi][ni], false, false);
    __syncthreads();
  }

#pragma unroll
  for (int mi = 0; mi < 2; ++mi)
#pragma unroll
    for (int ni = 0; ni < 2; ++ni) {
      const int col = wx * 32 + ni * 16 + l16;
      const float bzc = bz[col];
#pragma unroll
      for (int r = 0; r < 8; ++r) {
        const int row = mBlock + wy * 32 + mi * 16 + half * 8 + r;
        if (row < NN) z[(size_t)row * HD + col] = acc[mi][ni][r] + bzc;
      }
    }
}

// ---------------------------------------------------------------- gather 1 + BN + ReLU
// One wave per node: sum z[src] rows of the node's in-edges (float4 lanes),
// then aggbn = bf16(relu((sum + bg1 - rmean) * gamma*rsqrt(rvar+eps) + beta)).
__global__ __launch_bounds__(256) void agg1_bn_kernel(
    const int* __restrict__ rowStart, const int* __restrict__ csrSrc,
    const float* __restrict__ z,
    const float* __restrict__ bg1, const float* __restrict__ gamma,
    const float* __restrict__ beta, const float* __restrict__ rmean,
    const float* __restrict__ rvar, __bf16* __restrict__ aggbn) {
  const int gw   = (blockIdx.x * 256 + threadIdx.x) >> 5;
  const int lane = threadIdx.x & 31;
  if (gw >= NN) return;
  const int n  = gw;
  const int cb = lane * 4;

  float sc[4], sh[4];
#pragma unroll
  for (int j = 0; j < 4; ++j) {
    const int c = cb + j;
    const float s = gamma[c] * rsqrtf(rvar[c] + BNEPS);
    sc[j] = s;
    sh[j] = beta[c] + (bg1[c] - rmean[c]) * s;
  }

  const int beg = rowStart[n];
  const int end = rowStart[n + 1];
  float a0 = 0.f, a1 = 0.f, a2 = 0.f, a3 = 0.f;
  int j = beg;
  for (; j + 1 < end; j += 2) {          // 2 outstanding gathers per iter
    const int s0 = csrSrc[j];
    const int s1 = csrSrc[j + 1];
    const float4 v0 = *(const float4*)(z + (size_t)s0 * HD + cb);
    const float4 v1 = *(const float4*)(z + (size_t)s1 * HD + cb);
    a0 += v0.x + v1.x; a1 += v0.y + v1.y;
    a2 += v0.z + v1.z; a3 += v0.w + v1.w;
  }
  if (j < end) {
    const int s0 = csrSrc[j];
    const float4 v0 = *(const float4*)(z + (size_t)s0 * HD + cb);
    a0 += v0.x; a1 += v0.y; a2 += v0.z; a3 += v0.w;
  }

  __bf16* o = aggbn + (size_t)n * HD + cb;
  o[0] = (__bf16)fmaxf(a0 * sc[0] + sh[0], 0.f);
  o[1] = (__bf16)fmaxf(a1 * sc[1] + sh[1], 0.f);
  o[2] = (__bf16)fmaxf(a2 * sc[2] + sh[2], 0.f);
  o[3] = (__bf16)fmaxf(a3 * sc[3] + sh[3], 0.f);
}

// ---------------------------------------------------------------- GEMM2
// z2[N,64] = aggbn @ Bt2^T via bf16 WMMA; K=128, A already bf16 in global.
// Block: 8 waves as 4(row)x2(col); tile = 128 rows x 64 cols.
__global__ __launch_bounds__(256) void gemm2_kernel(
    const __bf16* __restrict__ A, const __bf16* __restrict__ Bt2,
    float* __restrict__ z2) {
  const int tid  = threadIdx.x;
  const int lane = tid & 31;
  const int w    = tid >> 5;
  const int wy   = w >> 1;       // 0..3
  const int wx   = w & 1;        // 0..1
  const int half = lane >> 4;
  const int l16  = lane & 15;
  const int mBlock = blockIdx.x * 128;

  v8f acc[2][2];
#pragma unroll
  for (int mi = 0; mi < 2; ++mi)
#pragma unroll
    for (int ni = 0; ni < 2; ++ni)
      acc[mi][ni] = (v8f){0.f, 0.f, 0.f, 0.f, 0.f, 0.f, 0.f, 0.f};

#pragma unroll
  for (int kk = 0; kk < HD; kk += 32) {
    v16bf a[2], b[2];
#pragma unroll
    for (int mi = 0; mi < 2; ++mi) {
      int row = mBlock + wy * 32 + mi * 16 + l16;
      if (row >= NN) row = NN - 1;                 // clamp; stores guarded
      const __bf16* p = A + (size_t)row * HD + kk + half * 16;
      a[mi] = load_a_frag(p);
    }
#pragma unroll
    for (int ni = 0; ni < 2; ++ni) {
      const __bf16* q = Bt2 + (size_t)(wx * 32 + ni * 16 + l16) * HD + kk + half * 16;
      b[ni] = *reinterpret_cast<const v16bf*>(q);
    }
#pragma unroll
    for (int mi = 0; mi < 2; ++mi)
#pragma unroll
      for (int ni = 0; ni < 2; ++ni)
        acc[mi][ni] = __builtin_amdgcn_wmma_f32_16x16x32_bf16(
            false, a[mi], false, b[ni], (short)0, acc[mi][ni], false, false);
  }

#pragma unroll
  for (int mi = 0; mi < 2; ++mi)
#pragma unroll
    for (int ni = 0; ni < 2; ++ni) {
      const int col = wx * 32 + ni * 16 + l16;
#pragma unroll
      for (int r = 0; r < 8; ++r) {
        const int row = mBlock + wy * 32 + mi * 16 + half * 8 + r;
        if (row < NN) z2[(size_t)row * CD + col] = acc[mi][ni][r];
      }
    }
}

// ---------------------------------------------------------------- gather 2
// Two nodes per wave (C=64 -> 16 float4 lanes each): out[n] = bg2 + sum z2[src].
__global__ __launch_bounds__(256) void agg2_kernel(
    const int* __restrict__ rowStart, const int* __restrict__ csrSrc,
    const float* __restrict__ z2, const float* __restrict__ bg2,
    float* __restrict__ out) {
  const int gw   = (blockIdx.x * 256 + threadIdx.x) >> 5;
  const int lane = threadIdx.x & 31;
  const int n    = gw * 2 + (lane >> 4);
  if (n >= NN) return;
  const int cb = (lane & 15) * 4;

  float a0 = bg2[cb], a1 = bg2[cb + 1], a2 = bg2[cb + 2], a3 = bg2[cb + 3];
  const int beg = rowStart[n];
  const int end = rowStart[n + 1];
  int j = beg;
  for (; j + 1 < end; j += 2) {
    const int s0 = csrSrc[j];
    const int s1 = csrSrc[j + 1];
    const float4 v0 = *(const float4*)(z2 + (size_t)s0 * CD + cb);
    const float4 v1 = *(const float4*)(z2 + (size_t)s1 * CD + cb);
    a0 += v0.x + v1.x; a1 += v0.y + v1.y;
    a2 += v0.z + v1.z; a3 += v0.w + v1.w;
  }
  if (j < end) {
    const int s0 = csrSrc[j];
    const float4 v0 = *(const float4*)(z2 + (size_t)s0 * CD + cb);
    a0 += v0.x; a1 += v0.y; a2 += v0.z; a3 += v0.w;
  }
  *(float4*)(out + (size_t)n * CD + cb) = make_float4(a0, a1, a2, a3);
}

// ---------------------------------------------------------------- launch
extern "C" void kernel_launch(void* const* d_in, const int* in_sizes, int n_in,
                              void* d_out, int out_size, void* d_ws, size_t ws_size,
                              hipStream_t stream) {
  (void)in_sizes; (void)n_in; (void)out_size; (void)ws_size;
  const float* x1    = (const float*)d_in[0];
  const float* x2    = (const float*)d_in[1];
  const int*   ei    = (const int*)d_in[2];   // JAX x64 off -> int32 on device
  const float* W1    = (const float*)d_in[3];
  const float* b1    = (const float*)d_in[4];
  const float* W2    = (const float*)d_in[5];
  const float* b2    = (const float*)d_in[6];
  const float* Wg1   = (const float*)d_in[7];
  const float* bg1   = (const float*)d_in[8];
  const float* Wg2   = (const float*)d_in[9];
  const float* bg2   = (const float*)d_in[10];
  const float* gamma = (const float*)d_in[11];
  const float* beta  = (const float*)d_in[12];
  const float* rmean = (const float*)d_in[13];
  const float* rvar  = (const float*)d_in[14];
  float* out = (float*)d_out;

  // workspace layout (all offsets 128B-aligned)
  char* ws = (char*)d_ws;
  __bf16* BtZ      = (__bf16*)(ws);                    //   163,840 B
  __bf16* Bt2      = (__bf16*)(ws + 163840);           //    16,384 B
  float*  bz       = (float*) (ws + 180224);           //       512 B
  int*    deg      = (int*)   (ws + 180736);           //   400,000 B
  int*    cursor   = (int*)   (ws + 580736);           //   400,000 B
  int*    rowStart = (int*)   (ws + 980736);           //   400,128 B (NN+1 ints)
  int*    csrSrc   = (int*)   (ws + 1380864);          // 6,400,000 B
  float*  z        = (float*) (ws + 7780864);          // 51,200,000 B (then aggbn bf16)
  float*  z2       = (float*) (ws + 58980864);         // 25,600,000 B
  __bf16* aggbn    = (__bf16*)z;                       // z dead after agg1

  prep_kernel    <<<353,   256,  0, stream>>>(W1, b1, W2, b2, Wg1, Wg2, BtZ, Bt2, bz);
  zero_deg_kernel<<<391,   256,  0, stream>>>(deg);
  deg_kernel     <<<6250,  256,  0, stream>>>(ei, deg);
  scan_kernel    <<<1,     1024, 0, stream>>>(deg, rowStart, cursor);
  csr_fill_kernel<<<6250,  256,  0, stream>>>(ei, cursor, csrSrc);
  gemm1_kernel   <<<1563,  256,  0, stream>>>(x1, x2, BtZ, bz, z);
  agg1_bn_kernel <<<12500, 256,  0, stream>>>(rowStart, csrSrc, z, bg1, gamma, beta,
                                              rmean, rvar, aggbn);
  gemm2_kernel   <<<782,   256,  0, stream>>>(aggbn, Bt2, z2);
  agg2_kernel    <<<6250,  256,  0, stream>>>(rowStart, csrSrc, z2, bg2, out);
}
